// HeteroGNN_52707838656539
// MI455X (gfx1250) — matrix operand order repaired and checked by reading, not compiled, not run on hardware
//
#include <hip/hip_runtime.h>

typedef __bf16 bf16;
typedef __attribute__((ext_vector_type(16))) __bf16 v16bf;
typedef __attribute__((ext_vector_type(8)))  float  v8f;

#define NA_   200000
#define NPAP_ 400000
#define NG_   512
#define EW_   2000000
#define EC_   2000000
#define ES_   200000

// ---------------- degree counting ----------------
__global__ void k_count(const int* __restrict__ dst, float* __restrict__ cnt, int n) {
  int i = blockIdx.x * blockDim.x + threadIdx.x;
  if (i < n) atomicAdd(&cnt[dst[i]], 1.0f);
}

__global__ void k_invert(float* __restrict__ c, int n) {
  int i = blockIdx.x * blockDim.x + threadIdx.x;
  if (i < n) c[i] = 1.0f / fmaxf(c[i], 1.0f);
}

// ------------- edge scatter: gather 64-f32 src row, atomic-add into dst row -------------
// one wave (32 lanes) per edge; each lane moves a float2 (cols 2*lane, 2*lane+1)
__global__ void k_scatter(const float* __restrict__ x, const int* __restrict__ src,
                          const int* __restrict__ dst, float* __restrict__ agg, int nedges) {
  long long g = (long long)blockIdx.x * blockDim.x + threadIdx.x;
  int e    = (int)(g >> 5);
  int lane = (int)(g & 31);
  if (e >= nedges) return;
  int s = src[e], d = dst[e];
  const float2 v = *(const float2*)(x + (size_t)s * 64 + lane * 2);
  float* o = agg + (size_t)d * 64 + lane * 2;
  atomicAdd(o, v.x);
  atomicAdd(o + 1, v.y);
}

// ------------- pre-swizzle weights into B-fragment order (bf16) -------------
// element order: idx = ((s*4 + nt)*32 + lane)*16 + e
// B layout (16b, 32x16): k = 32s + (lane>=16 ? 16 : 0) + e ; n = 16*nt + (lane&15)
// combined K axis: seg0 (k<64)=Wl0, seg1 (64..127)=Wr0(+Wr1), seg2 (128..191)=Wl2
__global__ void k_pack_w(const float* __restrict__ Wl0, const float* __restrict__ Wr0,
                         const float* __restrict__ Wr1, const float* __restrict__ Wl2,
                         unsigned short* __restrict__ outp, int ksteps) {
  int idx = blockIdx.x * blockDim.x + threadIdx.x;
  int total = (ksteps * 4 * 32) * 16;
  if (idx >= total) return;
  int e    = idx & 15;
  int lane = (idx >> 4) & 31;
  int nt   = (idx >> 9) & 3;
  int s    = idx >> 11;
  int k = s * 32 + ((lane >= 16) ? 16 : 0) + e;
  int n = nt * 16 + (lane & 15);
  int seg = k >> 6, kc = k & 63;
  float w;
  if (seg == 0)      w = Wl0[kc * 64 + n];
  else if (seg == 1) { w = Wr0[kc * 64 + n]; if (Wr1) w += Wr1[kc * 64 + n]; }
  else               w = Wl2[kc * 64 + n];
  bf16 b = (bf16)w;
  outp[idx] = __builtin_bit_cast(unsigned short, b);
}

// ------------- fused SAGE GEMM: Y = relu([agg0/deg0 | X | agg2/deg2] @ Wpack + bias) -------------
// wave32; one 16x64 output tile per wave; bf16 WMMA with f32 accumulation.
__global__ __launch_bounds__(256) void k_sage_gemm(
    const float* __restrict__ segA, const float* __restrict__ invA,
    const float* __restrict__ segX,
    const float* __restrict__ segC, const float* __restrict__ invC,
    const unsigned short* __restrict__ wpack,
    const float* __restrict__ bias0, const float* __restrict__ bias1,
    float* __restrict__ out, int ntiles, int ksteps) {
  __shared__ __align__(16) unsigned short ldsW[6 * 4 * 512];  // up to K=192: 24 KB

  const int tid = threadIdx.x;
  // stage pre-swizzled weights in LDS
  {
    const unsigned int* s = (const unsigned int*)wpack;
    unsigned int* d = (unsigned int*)ldsW;
    const int words = ksteps * 4 * 512 / 2;
    for (int i = tid; i < words; i += 256) d[i] = s[i];
  }
  __syncthreads();

  const int lane = tid & 31;
  const int wave = tid >> 5;
  const int tile = blockIdx.x * 8 + wave;
  if (tile >= ntiles) return;

  const int m0   = tile * 16;
  const int mrow = m0 + (lane & 15);
  const float scA = invA[mrow];
  const float scC = segC ? invC[mrow] : 0.0f;
  const int hi = (lane >= 16) ? 16 : 0;

  v8f acc[4] = {};

  for (int s = 0; s < ksteps; ++s) {
    // ---- A fragment: one contiguous group of 16 f32, scaled + cvt to bf16 ----
    const int seg = s >> 1;
    const float* sp; float sc;
    if (seg == 0)      { sp = segA; sc = scA;  }
    else if (seg == 1) { sp = segX; sc = 1.0f; }
    else               { sp = segC; sc = scC;  }
    const int col0 = (s * 32 + hi) & 63;
    const float* p = sp + (size_t)mrow * 64 + col0;
    float4 f0 = *(const float4*)(p);
    float4 f1 = *(const float4*)(p + 4);
    float4 f2 = *(const float4*)(p + 8);
    float4 f3 = *(const float4*)(p + 12);
    v16bf a;
    a[0]  = (bf16)(f0.x * sc); a[1]  = (bf16)(f0.y * sc);
    a[2]  = (bf16)(f0.z * sc); a[3]  = (bf16)(f0.w * sc);
    a[4]  = (bf16)(f1.x * sc); a[5]  = (bf16)(f1.y * sc);
    a[6]  = (bf16)(f1.z * sc); a[7]  = (bf16)(f1.w * sc);
    a[8]  = (bf16)(f2.x * sc); a[9]  = (bf16)(f2.y * sc);
    a[10] = (bf16)(f2.z * sc); a[11] = (bf16)(f2.w * sc);
    a[12] = (bf16)(f3.x * sc); a[13] = (bf16)(f3.y * sc);
    a[14] = (bf16)(f3.z * sc); a[15] = (bf16)(f3.w * sc);

    // ---- 4 B fragments from LDS, 4 WMMAs ----
    const unsigned short* wb = &ldsW[s * (4 * 512) + lane * 16];
    v16bf b0 = *(const v16bf*)(wb);
    v16bf b1 = *(const v16bf*)(wb + 512);
    v16bf b2 = *(const v16bf*)(wb + 1024);
    v16bf b3 = *(const v16bf*)(wb + 1536);
    acc[0] = __builtin_amdgcn_wmma_f32_16x16x32_bf16(false, a, false, b0, (short)0, acc[0], false, false);
    acc[1] = __builtin_amdgcn_wmma_f32_16x16x32_bf16(false, a, false, b1, (short)0, acc[1], false, false);
    acc[2] = __builtin_amdgcn_wmma_f32_16x16x32_bf16(false, a, false, b2, (short)0, acc[2], false, false);
    acc[3] = __builtin_amdgcn_wmma_f32_16x16x32_bf16(false, a, false, b3, (short)0, acc[3], false, false);
  }

  // ---- store D with fused bias + ReLU; row = m0 + r + (lane>=16?8:0), col = 16nt+(lane&15) ----
  const int nlo   = lane & 15;
  const int rbase = m0 + ((lane >= 16) ? 8 : 0);
#pragma unroll
  for (int nt = 0; nt < 4; ++nt) {
    const int col = nt * 16 + nlo;
    float bb = bias0[col] + (bias1 ? bias1[col] : 0.0f);
#pragma unroll
    for (int r = 0; r < 8; ++r) {
      out[(size_t)(rbase + r) * 64 + col] = fmaxf(acc[nt][r] + bb, 0.0f);
    }
  }
}

// ------------- per-graph mean pooling (sum + count via atomics) -------------
__global__ void k_pool(const float* __restrict__ x, const int* __restrict__ batch,
                       float* __restrict__ pool, float* __restrict__ cnt, int n) {
  long long g = (long long)blockIdx.x * blockDim.x + threadIdx.x;
  int i    = (int)(g >> 5);
  int lane = (int)(g & 31);
  if (i >= n) return;
  int b = batch[i];
  const float2 v = *(const float2*)(x + (size_t)i * 64 + lane * 2);
  float* o = pool + (size_t)b * 64 + lane * 2;
  atomicAdd(o, v.x);
  atomicAdd(o + 1, v.y);
  if (lane == 0) atomicAdd(&cnt[b], 1.0f);
}

// ------------- final MLP: [512,128] -> relu fc1 [128,64] -> fc2 [64,2] -------------
__global__ __launch_bounds__(64) void k_mlp(const float* __restrict__ pool_a, const float* __restrict__ cnt_a,
                                            const float* __restrict__ pool_p, const float* __restrict__ cnt_p,
                                            const float* __restrict__ W1, const float* __restrict__ b1,
                                            const float* __restrict__ W2, const float* __restrict__ b2,
                                            float* __restrict__ out) {
  const int row = blockIdx.x;
  const int tid = threadIdx.x;
  __shared__ float g[128];
  __shared__ float h[64];
  const float ia = 1.0f / fmaxf(cnt_a[row], 1.0f);
  const float ip = 1.0f / fmaxf(cnt_p[row], 1.0f);
  g[tid]      = pool_a[(size_t)row * 64 + tid] * ia;
  g[64 + tid] = pool_p[(size_t)row * 64 + tid] * ip;
  __syncthreads();
  float acc = b1[tid];
#pragma unroll 8
  for (int k = 0; k < 128; ++k) acc += g[k] * W1[k * 64 + tid];
  h[tid] = fmaxf(acc, 0.0f);
  __syncthreads();
  if (tid < 2) {
    float o = b2[tid];
#pragma unroll 8
    for (int k = 0; k < 64; ++k) o += h[k] * W2[k * 2 + tid];
    out[row * 2 + tid] = o;
  }
}

extern "C" void kernel_launch(void* const* d_in, const int* in_sizes, int n_in,
                              void* d_out, int out_size, void* d_ws, size_t ws_size,
                              hipStream_t stream) {
  (void)in_sizes; (void)n_in; (void)out_size; (void)ws_size;

  const float* x_author = (const float*)d_in[0];
  const float* x_paper  = (const float*)d_in[1];
  const int* ws = (const int*)d_in[2];
  const int* wd = (const int*)d_in[3];
  const int* cs = (const int*)d_in[4];
  const int* cd = (const int*)d_in[5];
  const int* ss = (const int*)d_in[6];
  const int* sd = (const int*)d_in[7];
  const int* batch_a = (const int*)d_in[8];
  const int* batch_p = (const int*)d_in[9];
  // params (dict insertion order): per layer l in {0,1}: writes_{Wl,Wr,b}, cites_{Wl,Wr,b}, self_{Wl,Wr,b}
  auto P = [&](int i) { return (const float*)d_in[10 + i]; };
  const float* fc1_W = P(18);
  const float* fc1_b = P(19);
  const float* fc2_W = P(20);
  const float* fc2_b = P(21);

  // ---- workspace layout ----
  char* wsb = (char*)d_ws;
  size_t off = 0;
  auto alloc = [&](size_t bytes) { void* p = wsb + off; off = (off + bytes + 255) & ~(size_t)255; return p; };
  float* agg_w = (float*)alloc((size_t)NPAP_ * 64 * 4);
  float* agg_c = (float*)alloc((size_t)NPAP_ * 64 * 4);
  float* agg_s = (float*)alloc((size_t)NA_ * 64 * 4);
  float* xp1   = (float*)alloc((size_t)NPAP_ * 64 * 4);
  float* xa1   = (float*)alloc((size_t)NA_ * 64 * 4);
  float* inv_w = (float*)alloc((size_t)NPAP_ * 4);
  float* inv_c = (float*)alloc((size_t)NPAP_ * 4);
  float* inv_s = (float*)alloc((size_t)NA_ * 4);
  unsigned short* wpackP = (unsigned short*)alloc(12288 * 2);
  unsigned short* wpackA = (unsigned short*)alloc(8192 * 2);
  float* pool_a = (float*)alloc((size_t)NG_ * 64 * 4);
  float* pool_p = (float*)alloc((size_t)NG_ * 64 * 4);
  float* cg_a   = (float*)alloc((size_t)NG_ * 4);
  float* cg_p   = (float*)alloc((size_t)NG_ * 4);

  const int T = 256;
  auto blk = [&](long long n) { return (unsigned)((n + T - 1) / T); };

  // ---- in-degree reciprocals (shared by both layers) ----
  hipMemsetAsync(inv_w, 0, (size_t)NPAP_ * 4, stream);
  hipMemsetAsync(inv_c, 0, (size_t)NPAP_ * 4, stream);
  hipMemsetAsync(inv_s, 0, (size_t)NA_ * 4, stream);
  k_count<<<blk(EW_), T, 0, stream>>>(wd, inv_w, EW_);
  k_count<<<blk(EC_), T, 0, stream>>>(cd, inv_c, EC_);
  k_count<<<blk(ES_), T, 0, stream>>>(sd, inv_s, ES_);
  k_invert<<<blk(NPAP_), T, 0, stream>>>(inv_w, NPAP_);
  k_invert<<<blk(NPAP_), T, 0, stream>>>(inv_c, NPAP_);
  k_invert<<<blk(NA_), T, 0, stream>>>(inv_s, NA_);

  const float* in_a = x_author;
  const float* in_p = x_paper;
  for (int l = 0; l < 2; ++l) {
    const float* Wl_w = P(l * 9 + 0); const float* Wr_w = P(l * 9 + 1); const float* b_w = P(l * 9 + 2);
    const float* Wl_c = P(l * 9 + 3); const float* Wr_c = P(l * 9 + 4); const float* b_c = P(l * 9 + 5);
    const float* Wl_s = P(l * 9 + 6); const float* Wr_s = P(l * 9 + 7); const float* b_s = P(l * 9 + 8);

    hipMemsetAsync(agg_w, 0, (size_t)NPAP_ * 64 * 4, stream);
    hipMemsetAsync(agg_c, 0, (size_t)NPAP_ * 64 * 4, stream);
    hipMemsetAsync(agg_s, 0, (size_t)NA_ * 64 * 4, stream);
    k_scatter<<<blk((long long)EW_ * 32), T, 0, stream>>>(in_a, ws, wd, agg_w, EW_);
    k_scatter<<<blk((long long)EC_ * 32), T, 0, stream>>>(in_p, cs, cd, agg_c, EC_);
    k_scatter<<<blk((long long)ES_ * 32), T, 0, stream>>>(in_a, ss, sd, agg_s, ES_);

    // papers: K = 192 -> [Wl_w | Wr_w + Wr_c | Wl_c]
    k_pack_w<<<blk(12288), T, 0, stream>>>(Wl_w, Wr_w, Wr_c, Wl_c, wpackP, 6);
    // authors: K = 128 -> [Wl_s | Wr_s]
    k_pack_w<<<blk(8192), T, 0, stream>>>(Wl_s, Wr_s, nullptr, nullptr, wpackA, 4);

    const int tilesP = NPAP_ / 16, tilesA = NA_ / 16;
    k_sage_gemm<<<(tilesP + 7) / 8, T, 0, stream>>>(agg_w, inv_w, in_p, agg_c, inv_c,
                                                    wpackP, b_w, b_c, xp1, tilesP, 6);
    k_sage_gemm<<<(tilesA + 7) / 8, T, 0, stream>>>(agg_s, inv_s, in_a, nullptr, nullptr,
                                                    wpackA, b_s, nullptr, xa1, tilesA, 4);
    in_a = xa1;  // layer 2 (and in-place update: each row read/written only by its own wave)
    in_p = xp1;
  }

  // ---- per-graph mean pooling ----
  hipMemsetAsync(pool_a, 0, (size_t)NG_ * 64 * 4, stream);
  hipMemsetAsync(pool_p, 0, (size_t)NG_ * 64 * 4, stream);
  hipMemsetAsync(cg_a, 0, (size_t)NG_ * 4, stream);
  hipMemsetAsync(cg_p, 0, (size_t)NG_ * 4, stream);
  k_pool<<<blk((long long)NA_ * 32), T, 0, stream>>>(xa1, batch_a, pool_a, cg_a, NA_);
  k_pool<<<blk((long long)NPAP_ * 32), T, 0, stream>>>(xp1, batch_p, pool_p, cg_p, NPAP_);

  // ---- final MLP ----
  k_mlp<<<NG_, 64, 0, stream>>>(pool_a, cg_a, pool_p, cg_p, fc1_W, fc1_b, fc2_W, fc2_b, (float*)d_out);
}
